// NonLocalBlock_28475633172876
// MI455X (gfx1250) — compile-verified, matrix operand order checked by reading, not run on hardware
//
#include <hip/hip_runtime.h>

#define C_CH 256
#define HC 128
#define NB 4
#define NPIX 4096

typedef __attribute__((ext_vector_type(16))) __bf16 v16bf;
typedef __attribute__((ext_vector_type(8)))  float  v8f;
typedef __attribute__((ext_vector_type(4)))  int    v4i;

#define WMMA_BF16(a, b, c) \
  __builtin_amdgcn_wmma_f32_16x16x32_bf16(false, (a), false, (b), (short)0, (c), false, false)

// ---------------------------------------------------------------------------
// gfx1250 async global->LDS copy (ASYNCcnt path) with graceful fallback.
// Builtin signature (from hipcc diagnostic): (v4i AS1*, v4i AS3*, imm, imm).
// ---------------------------------------------------------------------------
#if __has_builtin(__builtin_amdgcn_global_load_async_to_lds_b128)
#define HAVE_ASYNC_LDS 1
__device__ __forceinline__ void async_cp16(const void* g, void* l) {
  __builtin_amdgcn_global_load_async_to_lds_b128(
      (__attribute__((address_space(1))) v4i*)(g),
      (__attribute__((address_space(3))) v4i*)(l), 0, 0);
}
#else
#define HAVE_ASYNC_LDS 0
__device__ __forceinline__ void async_cp16(const void* g, void* l) {
  *(uint4*)l = *(const uint4*)g;
}
#endif

__device__ __forceinline__ void wait_async0() {
#if HAVE_ASYNC_LDS
#if __has_builtin(__builtin_amdgcn_s_wait_asynccnt)
  __builtin_amdgcn_s_wait_asynccnt(0);
#else
  asm volatile("s_wait_asynccnt 0" ::: "memory");
#endif
#endif
}

// ---------------------------------------------------------------------------
// Kernel 0: convert the four weight matrices fp32 -> bf16 (each 32768 elems)
// ---------------------------------------------------------------------------
__global__ __launch_bounds__(256) void prep_weights(
    const float* __restrict__ wt, const float* __restrict__ wp,
    const float* __restrict__ wg, const float* __restrict__ ww,
    __bf16* __restrict__ wtb, __bf16* __restrict__ wpb,
    __bf16* __restrict__ wgb, __bf16* __restrict__ wwb) {
  int i = blockIdx.x * 256 + threadIdx.x;   // 0 .. 131071
  int m = i >> 15;
  int k = i & 32767;
  const float* src = (m == 0) ? wt : (m == 1) ? wp : (m == 2) ? wg : ww;
  __bf16*      dst = (m == 0) ? wtb : (m == 1) ? wpb : (m == 2) ? wgb : wwb;
  dst[k] = (__bf16)src[k];
}

// ---------------------------------------------------------------------------
// Kernel 1: projections. theta/phi stored [B, N, HC]; g stored TRANSPOSED
// [B, HC, N] (via LDS transpose, coalesced b128 global stores) so the
// attention kernel can stage it with straight contiguous async copies.
// Grid: (N/64, B), block 256 (8 waves). Wave w: rows (w&3)*16, d-half (w>>2)*64.
// ---------------------------------------------------------------------------
#define XS_STRIDE 272   // 272*2 = 544 B = 17*32 -> 32B-aligned rows
#define GT_LSTRIDE 72   // 144 B rows (16B-aligned for uint4)
__global__ __launch_bounds__(256) void proj_kernel(
    const float* __restrict__ x,
    const __bf16* __restrict__ wtb, const float* __restrict__ bt,
    const __bf16* __restrict__ wpb, const float* __restrict__ bp,
    const __bf16* __restrict__ wgb, const float* __restrict__ bg,
    __bf16* __restrict__ theta, __bf16* __restrict__ phi,
    __bf16* __restrict__ gT) {
  __shared__ alignas(32) char smem[64 * XS_STRIDE * 2];   // 34816 B, reused
  __bf16 (*xs)[XS_STRIDE]  = (__bf16 (*)[XS_STRIDE])smem; // [64 n][256 c]
  __bf16 (*gt)[GT_LSTRIDE] = (__bf16 (*)[GT_LSTRIDE])smem;// [128 d][64 n]

  const int b   = blockIdx.y;
  const int n0  = blockIdx.x * 64;
  const int tid = threadIdx.x;
  const float* xb = x + (size_t)b * C_CH * NPIX;

  // Stage x^T block: 64 positions x 256 channels, coalesced along n.
  const int nloc  = tid & 63;
  const int cbase = tid >> 6;            // 0..3
  #pragma unroll 4
  for (int k = 0; k < 64; ++k) {
    int c = cbase + k * 4;
    xs[nloc][c] = (__bf16)xb[(size_t)c * NPIX + n0 + nloc];
  }
  __syncthreads();

  const int lane  = tid & 31;
  const int wave  = tid >> 5;
  const int rbase = (wave & 3) * 16;
  const int dbase = (wave >> 2) * 64;
  const int arow  = rbase + (lane & 15);
  const int lo16  = lane & 15;
  const int hi16  = lane >> 4;

  // --- theta and phi: straight [B, N, HC] outputs ---
  const __bf16* wmat[2] = {wtb, wpb};
  const float*  bias[2] = {bt, bp};
  __bf16*       outp[2] = {theta, phi};
  #pragma unroll
  for (int m = 0; m < 2; ++m) {
    v8f acc[4] = {};
    #pragma unroll
    for (int kk = 0; kk < 8; ++kk) {            // K = 256
      const int k0 = kk * 32 + hi16 * 16;
      v16bf a = *(const v16bf*)&xs[arow][k0];
      #pragma unroll
      for (int t = 0; t < 4; ++t) {
        const int d = dbase + t * 16 + lo16;
        v16bf bfr = *(const v16bf*)(wmat[m] + (size_t)d * C_CH + k0);
        acc[t] = WMMA_BF16(a, bfr, acc[t]);
      }
    }
    #pragma unroll
    for (int t = 0; t < 4; ++t) {
      const int d  = dbase + t * 16 + lo16;
      const float bv = bias[m][d];
      #pragma unroll
      for (int r = 0; r < 8; ++r) {
        const int nrow = n0 + rbase + r + hi16 * 8;
        outp[m][((size_t)b * NPIX + nrow) * HC + d] = (__bf16)(acc[t][r] + bv);
      }
    }
  }

  // --- g: compute, then transpose through LDS, store [B, HC, N] ---
  v8f gacc[4] = {};
  #pragma unroll
  for (int kk = 0; kk < 8; ++kk) {
    const int k0 = kk * 32 + hi16 * 16;
    v16bf a = *(const v16bf*)&xs[arow][k0];
    #pragma unroll
    for (int t = 0; t < 4; ++t) {
      const int d = dbase + t * 16 + lo16;
      v16bf bfr = *(const v16bf*)(wgb + (size_t)d * C_CH + k0);
      gacc[t] = WMMA_BF16(a, bfr, gacc[t]);
    }
  }
  float gbv[4];
  #pragma unroll
  for (int t = 0; t < 4; ++t) gbv[t] = bg[dbase + t * 16 + lo16];

  __syncthreads();   // all waves done reading xs before aliased gt is written
  #pragma unroll
  for (int t = 0; t < 4; ++t) {
    const int d = dbase + t * 16 + lo16;
    #pragma unroll
    for (int r = 0; r < 8; ++r)
      gt[d][rbase + r + hi16 * 8] = (__bf16)(gacc[t][r] + gbv[t]);
  }
  __syncthreads();
  // cooperative coalesced store of the 128 x 64 transposed tile
  __bf16* gTb = gT + (size_t)b * HC * NPIX;
  #pragma unroll
  for (int k = 0; k < 4; ++k) {
    int idx = tid + k * 256;                 // 1024 chunks of 8 bf16
    int d   = idx >> 3;
    int ch  = idx & 7;
    *(uint4*)(gTb + (size_t)d * NPIX + n0 + ch * 8) =
        *(const uint4*)&gt[d][ch * 8];
  }
}

// ---------------------------------------------------------------------------
// Kernel 2: fused flash attention with double-buffered async LDS staging.
// Grid: (N/128, B), block 256 (8 waves, 16 query rows each).
// ---------------------------------------------------------------------------
#define PH_STRIDE 144   // 288 B rows
#define GT_STRIDE 80    // 160 B rows
#define NJ (NPIX / 64)
__global__ __launch_bounds__(256) void attn_kernel(
    const __bf16* __restrict__ theta, const __bf16* __restrict__ phi,
    const __bf16* __restrict__ gT, __bf16* __restrict__ y) {
  __shared__ alignas(32) __bf16 phs[2][64][PH_STRIDE];   // 36864 B
  __shared__ alignas(32) __bf16 gts[2][HC][GT_STRIDE];   // 40960 B
  __shared__ alignas(32) __bf16 ps[8][16][GT_STRIDE];    // 20480 B

  const int b    = blockIdx.y;
  const int q0   = blockIdx.x * 128;
  const int tid  = threadIdx.x;
  const int lane = tid & 31;
  const int wave = tid >> 5;
  const int lo16 = lane & 15;
  const int hi16 = lane >> 4;

  const __bf16* thb = theta + (size_t)b * NPIX * HC;
  const __bf16* phb = phi   + (size_t)b * NPIX * HC;
  const __bf16* gtb = gT    + (size_t)b * HC * NPIX;

  const int qrow = q0 + wave * 16 + lo16;     // A-fragment row for this lane

  // Hoist theta A fragments: invariant across all 64 column blocks.
  v16bf afrag[4];
  #pragma unroll
  for (int kk = 0; kk < 4; ++kk)
    afrag[kk] = *(const v16bf*)(thb + (size_t)qrow * HC + kk * 32 + hi16 * 16);

  auto stage = [&](int j, int buf) {
    const int cb = j * 64;
    #pragma unroll
    for (int k = 0; k < 4; ++k) {            // phi block [64][128]
      int idx = tid + k * 256;
      int row = idx >> 4, ch = idx & 15;
      async_cp16(phb + (size_t)(cb + row) * HC + ch * 8, &phs[buf][row][ch * 8]);
    }
    #pragma unroll
    for (int k = 0; k < 4; ++k) {            // gT block [128][64], contiguous
      int idx = tid + k * 256;
      int d = idx >> 3, ch = idx & 7;
      async_cp16(gtb + (size_t)d * NPIX + cb + ch * 8, &gts[buf][d][ch * 8]);
    }
  };

  float m_st[8], l_st[8];
  #pragma unroll
  for (int r = 0; r < 8; ++r) { m_st[r] = -1e30f; l_st[r] = 0.0f; }
  v8f oacc[8] = {};

  stage(0, 0);
  wait_async0();
  __syncthreads();

  for (int j = 0; j < NJ; ++j) {
    const int cur = j & 1, nxt = cur ^ 1;
    if (j + 1 < NJ) stage(j + 1, nxt);       // overlaps with compute below

    // --- S = theta_tile (16x128) @ phi_block^T (128x64) ---
    v8f sacc[4] = {};
    #pragma unroll
    for (int kk = 0; kk < 4; ++kk) {
      const int k0 = kk * 32 + hi16 * 16;
      #pragma unroll
      for (int t = 0; t < 4; ++t) {
        v16bf bfr = *(const v16bf*)&phs[cur][t * 16 + lo16][k0];
        sacc[t] = WMMA_BF16(afrag[kk], bfr, sacc[t]);
      }
    }

    // --- online softmax (row stats reduced over the 16-lane row group) ---
    #pragma unroll
    for (int r = 0; r < 8; ++r) {
      float mb = fmaxf(fmaxf(sacc[0][r], sacc[1][r]),
                       fmaxf(sacc[2][r], sacc[3][r]));
      #pragma unroll
      for (int msk = 8; msk >= 1; msk >>= 1)
        mb = fmaxf(mb, __shfl_xor(mb, msk, 32));
      const float mnew  = fmaxf(m_st[r], mb);
      const float alpha = __expf(m_st[r] - mnew);
      m_st[r] = mnew;
      float rs = 0.0f;
      #pragma unroll
      for (int t = 0; t < 4; ++t) {
        sacc[t][r] = __expf(sacc[t][r] - mnew);
        rs += sacc[t][r];
      }
      #pragma unroll
      for (int msk = 8; msk >= 1; msk >>= 1)
        rs += __shfl_xor(rs, msk, 32);
      l_st[r] = l_st[r] * alpha + rs;
      #pragma unroll
      for (int t = 0; t < 8; ++t) oacc[t][r] *= alpha;
      const int prow = r + hi16 * 8;
      #pragma unroll
      for (int t = 0; t < 4; ++t)
        ps[wave][prow][t * 16 + lo16] = (__bf16)sacc[t][r];
    }

    // --- O += P (16x64) @ g_block (64x128) ---
    #pragma unroll
    for (int kk = 0; kk < 2; ++kk) {
      const int k0 = kk * 32 + hi16 * 16;
      v16bf a = *(const v16bf*)&ps[wave][lo16][k0];
      #pragma unroll
      for (int t = 0; t < 8; ++t) {
        v16bf bfr = *(const v16bf*)&gts[cur][t * 16 + lo16][k0];
        oacc[t] = WMMA_BF16(a, bfr, oacc[t]);
      }
    }

    wait_async0();      // prefetch of block j+1 landed
    __syncthreads();    // and everyone is done reading buffer `cur`
  }

  // --- finalize: y = O / l ---
  #pragma unroll
  for (int r = 0; r < 8; ++r) {
    const float inv  = 1.0f / l_st[r];
    const int   nrow = q0 + wave * 16 + r + hi16 * 8;
    #pragma unroll
    for (int t = 0; t < 8; ++t)
      y[((size_t)b * NPIX + nrow) * HC + t * 16 + lo16] =
          (__bf16)(oacc[t][r] * inv);
  }
}

// ---------------------------------------------------------------------------
// Kernel 3: z[b,o,n] = sum_d w_w[o,d] y[b,n,d] + b_w[o] + x[b,o,n]
// Grid: (N/128, C/128, B), block 256.
// ---------------------------------------------------------------------------
__global__ __launch_bounds__(256) void zout_kernel(
    const __bf16* __restrict__ y, const __bf16* __restrict__ wwb,
    const float* __restrict__ bw, const float* __restrict__ x,
    float* __restrict__ out) {
  const int b    = blockIdx.z;
  const int ob   = blockIdx.y;
  const int nb   = blockIdx.x;
  const int tid  = threadIdx.x;
  const int lane = tid & 31;
  const int wave = tid >> 5;
  const int lo16 = lane & 15;
  const int hi16 = lane >> 4;

  const int o0 = ob * 128 + wave * 16;
  const __bf16* yb = y + (size_t)b * NPIX * HC;

  v8f acc[8] = {};
  #pragma unroll
  for (int kk = 0; kk < 4; ++kk) {             // K = HC = 128
    const int k0 = kk * 32 + hi16 * 16;
    v16bf a = *(const v16bf*)(wwb + (size_t)(o0 + lo16) * HC + k0);
    #pragma unroll
    for (int t = 0; t < 8; ++t) {
      const int n = nb * 128 + t * 16 + lo16;
      v16bf bfr = *(const v16bf*)(yb + (size_t)n * HC + k0);
      acc[t] = WMMA_BF16(a, bfr, acc[t]);
    }
  }
  #pragma unroll
  for (int r = 0; r < 8; ++r) {
    const int  orow = o0 + r + hi16 * 8;
    const float bias = bw[orow];
    #pragma unroll
    for (int t = 0; t < 8; ++t) {
      const int    n   = nb * 128 + t * 16 + lo16;
      const size_t off = ((size_t)b * C_CH + orow) * NPIX + n;
      out[off] = acc[t][r] + bias + x[off];
    }
  }
}

// ---------------------------------------------------------------------------
extern "C" void kernel_launch(void* const* d_in, const int* in_sizes, int n_in,
                              void* d_out, int out_size, void* d_ws,
                              size_t ws_size, hipStream_t stream) {
  (void)in_sizes; (void)n_in; (void)out_size; (void)ws_size;
  const float* x       = (const float*)d_in[0];
  const float* w_theta = (const float*)d_in[1];
  const float* b_theta = (const float*)d_in[2];
  const float* w_phi   = (const float*)d_in[3];
  const float* b_phi   = (const float*)d_in[4];
  const float* w_g     = (const float*)d_in[5];
  const float* b_g     = (const float*)d_in[6];
  const float* w_w     = (const float*)d_in[7];
  const float* b_w     = (const float*)d_in[8];
  float* out = (float*)d_out;

  char* ws = (char*)d_ws;
  const size_t WSZ = (size_t)HC * C_CH * 2;           // 65536 B per weight
  const size_t TSZ = (size_t)NB * NPIX * HC * 2;      // 4 MiB per activation
  __bf16* wtb   = (__bf16*)(ws);
  __bf16* wpb   = (__bf16*)(ws + WSZ);
  __bf16* wgb   = (__bf16*)(ws + 2 * WSZ);
  __bf16* wwb   = (__bf16*)(ws + 3 * WSZ);
  __bf16* theta = (__bf16*)(ws + 4 * WSZ);
  __bf16* phi   = (__bf16*)(ws + 4 * WSZ + TSZ);
  __bf16* gTbuf = (__bf16*)(ws + 4 * WSZ + 2 * TSZ);  // [B, HC, N]
  __bf16* ybuf  = (__bf16*)(ws + 4 * WSZ + 3 * TSZ);

  prep_weights<<<512, 256, 0, stream>>>(w_theta, w_phi, w_g, w_w,
                                        wtb, wpb, wgb, wwb);

  dim3 gp(NPIX / 64, NB);
  proj_kernel<<<gp, 256, 0, stream>>>(x, wtb, b_theta, wpb, b_phi, wgb, b_g,
                                      theta, phi, gTbuf);

  dim3 ga(NPIX / 128, NB);
  attn_kernel<<<ga, 256, 0, stream>>>(theta, phi, gTbuf, ybuf);

  dim3 gz(NPIX / 128, C_CH / 128, NB);
  zout_kernel<<<gz, 256, 0, stream>>>(ybuf, wwb, b_w, x, out);
}
